// NeuralMJP_87162066305759
// MI455X (gfx1250) — matrix operand order repaired and checked by reading, not compiled, not run on hardware
//
#include <hip/hip_runtime.h>
#include <hip/hip_bf16.h>

typedef __attribute__((ext_vector_type(16))) _Float16 v16h;
typedef __attribute__((ext_vector_type(8)))  _Float16 h8;
typedef __attribute__((ext_vector_type(8)))  float    v8f;

#define B_ROWS 524288
#define DIM 32
#define HID 128
#define NS 32
#define WAVES 8
#define TILE 16

// LDS: weights (pre-swizzled f16 A-fragments) + biases only. Activations live in registers.
#define FRAG_BYTES 65536            // 64 frags * 32 lanes * 16 halves * 2B
#define BIAS_BYTES 2560             // 640 floats
#define SMEM_TOTAL (FRAG_BYTES + BIAS_BYTES)   // 68096 B -> 4 workgroups / WGP

__device__ __forceinline__ v8f wmma16(v16h a, v16h b, v8f c) {
    return __builtin_amdgcn_wmma_f32_16x16x32_f16(false, a, false, b, (short)0, c, false, false);
}

// Pre-swizzled A fragment: [lane][16 halves] contiguous -> two ds_load_b128
__device__ __forceinline__ v16h load_frag(const _Float16* p) {
    h8 lo = *(const h8*)p;
    h8 hi = *(const h8*)(p + 8);
    return __builtin_shufflevector(lo, hi, 0,1,2,3,4,5,6,7,8,9,10,11,12,13,14,15);
}

__device__ __forceinline__ uint32_t pack2h(float a, float b) {
    union { _Float16 h[2]; uint32_t u; } t;
    t.h[0] = (_Float16)a; t.h[1] = (_Float16)b;
    return t.u;
}

union BFrag { uint32_t u[8]; v16h v; };

// Half-wave transpose: D-tiles of pair (j0=2kk, j1=2kk+1), packed per lane as
// p0[4]/p1[4] (8 f16 each), into the B-fragment for K-block kk of next layer.
// Dest lane (kh) takes j = 2kk+kh: lo 8 halves from kh'=0 lanes, hi from kh'=1.
__device__ __forceinline__ void xpose_pair(const uint32_t p0[4], const uint32_t p1[4],
                                           int kh, uint32_t outp[8]) {
    #pragma unroll
    for (int q = 0; q < 4; ++q) {
        uint32_t x0 = __shfl_xor(p0[q], 16, 32);
        uint32_t x1 = __shfl_xor(p1[q], 16, 32);
        outp[q]     = kh ? x1    : p0[q];
        outp[4 + q] = kh ? p1[q] : x0;
    }
}

__global__ void __launch_bounds__(256) neuralmjp_fused(
    const float* __restrict__ hin, const float* __restrict__ uin,
    const float* __restrict__ qw1, const float* __restrict__ qb1,
    const float* __restrict__ qw2, const float* __restrict__ qb2,
    const float* __restrict__ giw1, const float* __restrict__ gib1,
    const float* __restrict__ giw2, const float* __restrict__ gib2,
    const float* __restrict__ gow1, const float* __restrict__ gob1,
    const float* __restrict__ gow2, const float* __restrict__ gob2,
    const float* __restrict__ dw1, const float* __restrict__ db1v,
    const float* __restrict__ dw2, const float* __restrict__ db2v,
    float* __restrict__ out)
{
    extern __shared__ char smem[];
    _Float16* frag  = (_Float16*)smem;
    float*    biasL = (float*)(smem + FRAG_BYTES);

    const int tid  = threadIdx.x;
    const int wave = tid >> 5;
    const int lane = tid & 31;

    // ---------------- init: weights -> transposed WMMA A-fragments ----------------
    // A-layout, K within 32-block: K=16*kh'+e
    for (int i = tid; i < 64 * 512; i += 256) {
        int f   = i >> 9;
        int rem = i & 511;
        int fl  = rem >> 4;
        int e   = rem & 15;
        int ml  = fl & 15;                                  // A row within 16-tile
        int Kp  = 16 * (fl >> 4) + e;                       // K within 32-block
        float val;
        if (f < 24) {                       // W1^T per head: A row = hidden, K = input d
            int t = f >> 3, j = f & 7;
            const float* w = (t == 0) ? qw1 : (t == 1) ? giw1 : gow1;   // [32][128]
            val = w[Kp * HID + j * 16 + ml];
        } else if (f < 48) {                // W2^T per head: A row = state, K = hidden
            int g = f - 24; int t = g >> 3; int r = g & 7; int jn = r >> 2; int kk = r & 3;
            const float* w = (t == 0) ? qw2 : (t == 1) ? giw2 : gow2;   // [128][32]
            val = w[(kk * 32 + Kp) * NS + jn * 16 + ml];
        } else if (f < 56) {                // dec_w1^T: A row = hidden, K = state
            int j = f - 48;
            val = dw1[Kp * HID + j * 16 + ml];                          // [32][128]
        } else {                            // dec_w2^T: A row = output d, K = hidden
            int r = f - 56; int jn = r >> 2; int kk = r & 3;
            val = dw2[(kk * 32 + Kp) * NS + jn * 16 + ml];              // [128][32]
        }
        frag[i] = (_Float16)val;
    }
    for (int i = tid; i < 640; i += 256) {
        float v;
        if (i < 384)      { int t = i >> 7; int n = i & 127; v = (t == 0 ? qb1 : t == 1 ? gib1 : gob1)[n]; }
        else if (i < 480) { int t = (i - 384) >> 5; int n = (i - 384) & 31; v = (t == 0 ? qb2 : t == 1 ? gib2 : gob2)[n]; }
        else if (i < 608) v = db1v[i - 480];
        else              v = db2v[i - 608];
        biasL[i] = v;
    }
    __syncthreads();    // LDS read-only from here on; no further barriers/fences needed

    const int tile = blockIdx.x * WAVES + wave;
    const int row0 = tile * TILE;
    const int m    = lane & 15;       // batch row within tile (B/D column of all tiles)
    const int kh   = lane >> 4;       // half-wave id

    // ---- h^T B-fragment: lane holds batch row m, K(d) = 16*kh + e, e=0..15 ----
    BFrag b_h;
    {
        const float* hr = hin + (size_t)(row0 + m) * DIM + 16 * kh;
        float4 p0 = *(const float4*)(hr);
        float4 p1 = *(const float4*)(hr + 4);
        float4 p2 = *(const float4*)(hr + 8);
        float4 p3 = *(const float4*)(hr + 12);
        b_h.u[0] = pack2h(p0.x, p0.y); b_h.u[1] = pack2h(p0.z, p0.w);
        b_h.u[2] = pack2h(p1.x, p1.y); b_h.u[3] = pack2h(p1.z, p1.w);
        b_h.u[4] = pack2h(p2.x, p2.y); b_h.u[5] = pack2h(p2.z, p2.w);
        b_h.u[6] = pack2h(p3.x, p3.y); b_h.u[7] = pack2h(p3.z, p3.w);
    }

    float prob[3][16];   // per-head softmax probs; lane's 16 states:
                         // i<8 -> s = 8*kh+i ; i>=8 -> s = 16+8*kh+(i-8)

    // ---------------- three MLP-softmax heads (all transposed) ----------------
    #pragma unroll
    for (int t = 0; t < 3; ++t) {
        // layer1: act^T tiles = W1^T(tile j) x h^T ; D: lane col=m, rows hid=16j+8kh+r
        BFrag bf2[4];
        #pragma unroll
        for (int kk = 0; kk < 4; ++kk) {
            uint32_t p0[4], p1[4];
            #pragma unroll
            for (int half = 0; half < 2; ++half) {
                int j = 2 * kk + half;
                v16h af = load_frag(frag + (size_t)(t * 8 + j) * 512 + lane * 16);
                v8f c = {};
                c = wmma16(af, b_h.v, c);
                const float* bb = biasL + t * 128 + j * 16 + 8 * kh;
                float4 bL = *(const float4*)(bb);
                float4 bH = *(const float4*)(bb + 4);
                float v0 = fmaxf(c[0] + bL.x, 0.f), v1 = fmaxf(c[1] + bL.y, 0.f);
                float v2 = fmaxf(c[2] + bL.z, 0.f), v3 = fmaxf(c[3] + bL.w, 0.f);
                float v4 = fmaxf(c[4] + bH.x, 0.f), v5 = fmaxf(c[5] + bH.y, 0.f);
                float v6 = fmaxf(c[6] + bH.z, 0.f), v7 = fmaxf(c[7] + bH.w, 0.f);
                uint32_t* p = half ? p1 : p0;
                p[0] = pack2h(v0, v1); p[1] = pack2h(v2, v3);
                p[2] = pack2h(v4, v5); p[3] = pack2h(v6, v7);
            }
            xpose_pair(p0, p1, kh, bf2[kk].u);
        }
        // layer2: logits^T = W2^T x act^T (K=128 -> 4 blocks, 2 state tiles)
        v8f c20 = {}, c21 = {};
        #pragma unroll
        for (int kk = 0; kk < 4; ++kk) {
            v16h a0 = load_frag(frag + (size_t)(24 + t * 8 + 0 + kk) * 512 + lane * 16);
            v16h a1 = load_frag(frag + (size_t)(24 + t * 8 + 4 + kk) * 512 + lane * 16);
            c20 = wmma16(a0, bf2[kk].v, c20);
            c21 = wmma16(a1, bf2[kk].v, c21);
        }
        // bias + in-register softmax (16 states in-lane + half-wave reduce)
        float L[16];
        {
            const float* bb = biasL + 384 + t * 32;
            float4 s0 = *(const float4*)(bb + 8 * kh);
            float4 s1 = *(const float4*)(bb + 8 * kh + 4);
            float4 s2 = *(const float4*)(bb + 16 + 8 * kh);
            float4 s3 = *(const float4*)(bb + 16 + 8 * kh + 4);
            L[0] = c20[0] + s0.x; L[1] = c20[1] + s0.y; L[2] = c20[2] + s0.z; L[3] = c20[3] + s0.w;
            L[4] = c20[4] + s1.x; L[5] = c20[5] + s1.y; L[6] = c20[6] + s1.z; L[7] = c20[7] + s1.w;
            L[8]  = c21[0] + s2.x; L[9]  = c21[1] + s2.y; L[10] = c21[2] + s2.z; L[11] = c21[3] + s2.w;
            L[12] = c21[4] + s3.x; L[13] = c21[5] + s3.y; L[14] = c21[6] + s3.z; L[15] = c21[7] + s3.w;
        }
        float mx = L[0];
        #pragma unroll
        for (int i = 1; i < 16; ++i) mx = fmaxf(mx, L[i]);
        mx = fmaxf(mx, __shfl_xor(mx, 16, 32));
        float ss = 0.f;
        #pragma unroll
        for (int i = 0; i < 16; ++i) { float e2 = __expf(L[i] - mx); L[i] = e2; ss += e2; }
        ss += __shfl_xor(ss, 16, 32);
        float inv = 1.f / ss;
        #pragma unroll
        for (int i = 0; i < 16; ++i) prob[t][i] = L[i] * inv;
    }

    // ---------------- master equation + gumbel straight-through (register-resident) ----------------
    BFrag b_s;
    {
        float qsum = 0.f;
        #pragma unroll
        for (int i = 0; i < 16; ++i) qsum += prob[0][i];
        qsum += __shfl_xor(qsum, 16, 32);

        const float* ur = uin + (size_t)(row0 + m) * NS;
        float4 u0 = *(const float4*)(ur + 8 * kh);
        float4 u1 = *(const float4*)(ur + 8 * kh + 4);
        float4 u2 = *(const float4*)(ur + 16 + 8 * kh);
        float4 u3 = *(const float4*)(ur + 16 + 8 * kh + 4);
        float uu[16] = { u0.x,u0.y,u0.z,u0.w, u1.x,u1.y,u1.z,u1.w,
                         u2.x,u2.y,u2.z,u2.w, u3.x,u3.y,u3.z,u3.w };

        float z[16];
        float zb = -3.4e38f; int sb = 0x7fffffff;
        #pragma unroll
        for (int i = 0; i < 16; ++i) {
            float qi = prob[0][i];
            float qn = qi + (prob[1][i] * (qsum - qi) + prob[2][i] * qi);
            float gn = -__logf(-__logf(uu[i] + 1e-20f) + 1e-20f);
            float zi = (__logf(qn + 1e-12f) + gn) / 1.0f;        // TAU = 1
            z[i] = zi;
            int s = (i < 8) ? (8 * kh + i) : (16 + 8 * kh + (i - 8));
            if (zi > zb) { zb = zi; sb = s; }                    // in-lane scan is s-ascending
        }
        {   // half-wave argmax merge (first-max tie break on smaller s)
            float zo = __shfl_xor(zb, 16, 32);
            int   so = __shfl_xor(sb, 16, 32);
            bool take = (zo > zb) || (zo == zb && so < sb);
            zb = take ? zo : zb;
            sb = take ? so : sb;
        }
        float es = 0.f;
        #pragma unroll
        for (int i = 0; i < 16; ++i) es += __expf(z[i] - zb);
        es += __shfl_xor(es, 16, 32);
        float inv = 1.f / es;

        uint32_t P0[4], P1[4];
        #pragma unroll
        for (int q = 0; q < 8; ++q) {
            int i0 = 2 * q, i1 = 2 * q + 1;
            int s0 = (i0 < 8) ? (8 * kh + i0) : (16 + 8 * kh + (i0 - 8));
            int s1 = (i1 < 8) ? (8 * kh + i1) : (16 + 8 * kh + (i1 - 8));
            float y0 = __expf(z[i0] - zb) * inv;
            float y1 = __expf(z[i1] - zb) * inv;
            float sv0 = (((s0 == sb) ? 1.f : 0.f) - y0) + y0;    // straight-through value
            float sv1 = (((s1 == sb) ? 1.f : 0.f) - y1) + y1;
            uint32_t pk = pack2h(sv0, sv1);
            if (q < 4) P0[q] = pk; else P1[q - 4] = pk;
        }
        xpose_pair(P0, P1, kh, b_s.u);   // sample^T B-fragment (K = 32 states)
    }

    // ---------------- decoder (transposed) ----------------
    BFrag bfd[4];
    #pragma unroll
    for (int kk = 0; kk < 4; ++kk) {
        uint32_t p0[4], p1[4];
        #pragma unroll
        for (int half = 0; half < 2; ++half) {
            int j = 2 * kk + half;
            v16h af = load_frag(frag + (size_t)(48 + j) * 512 + lane * 16);
            v8f c = {};
            c = wmma16(af, b_s.v, c);
            const float* bb = biasL + 480 + j * 16 + 8 * kh;
            float4 bL = *(const float4*)(bb);
            float4 bH = *(const float4*)(bb + 4);
            float v0 = fmaxf(c[0] + bL.x, 0.f), v1 = fmaxf(c[1] + bL.y, 0.f);
            float v2 = fmaxf(c[2] + bL.z, 0.f), v3 = fmaxf(c[3] + bL.w, 0.f);
            float v4 = fmaxf(c[4] + bH.x, 0.f), v5 = fmaxf(c[5] + bH.y, 0.f);
            float v6 = fmaxf(c[6] + bH.z, 0.f), v7 = fmaxf(c[7] + bH.w, 0.f);
            uint32_t* p = half ? p1 : p0;
            p[0] = pack2h(v0, v1); p[1] = pack2h(v2, v3);
            p[2] = pack2h(v4, v5); p[3] = pack2h(v6, v7);
        }
        xpose_pair(p0, p1, kh, bfd[kk].u);
    }
    v8f d0 = {}, d1 = {};
    #pragma unroll
    for (int kk = 0; kk < 4; ++kk) {
        v16h a0 = load_frag(frag + (size_t)(56 + 0 + kk) * 512 + lane * 16);
        v16h a1 = load_frag(frag + (size_t)(56 + 4 + kk) * 512 + lane * 16);
        d0 = wmma16(a0, bfd[kk].v, d0);
        d1 = wmma16(a1, bfd[kk].v, d1);
    }
    // out^T D-tiles: lane col = batch m, rows d = 16*jn + 8*kh + r  -> two contiguous
    // 8-float runs per lane -> direct global_store_b128 pairs, no staging.
    {
        float* orow = out + (size_t)(row0 + m) * DIM;
        const float* bb = biasL + 608;
        float4 s0 = *(const float4*)(bb + 8 * kh);
        float4 s1 = *(const float4*)(bb + 8 * kh + 4);
        float4 s2 = *(const float4*)(bb + 16 + 8 * kh);
        float4 s3 = *(const float4*)(bb + 16 + 8 * kh + 4);
        float4 o0 = { d0[0] + s0.x, d0[1] + s0.y, d0[2] + s0.z, d0[3] + s0.w };
        float4 o1 = { d0[4] + s1.x, d0[5] + s1.y, d0[6] + s1.z, d0[7] + s1.w };
        float4 o2 = { d1[0] + s2.x, d1[1] + s2.y, d1[2] + s2.z, d1[3] + s2.w };
        float4 o3 = { d1[4] + s3.x, d1[5] + s3.y, d1[6] + s3.z, d1[7] + s3.w };
        *(float4*)(orow + 8 * kh)          = o0;
        *(float4*)(orow + 8 * kh + 4)      = o1;
        *(float4*)(orow + 16 + 8 * kh)     = o2;
        *(float4*)(orow + 16 + 8 * kh + 4) = o3;
    }
}

extern "C" void kernel_launch(void* const* d_in, const int* in_sizes, int n_in,
                              void* d_out, int out_size, void* d_ws, size_t ws_size,
                              hipStream_t stream) {
    const float* h    = (const float*)d_in[0];
    const float* u    = (const float*)d_in[1];
    const float* qw1  = (const float*)d_in[2];
    const float* qb1  = (const float*)d_in[3];
    const float* qw2  = (const float*)d_in[4];
    const float* qb2  = (const float*)d_in[5];
    const float* giw1 = (const float*)d_in[6];
    const float* gib1 = (const float*)d_in[7];
    const float* giw2 = (const float*)d_in[8];
    const float* gib2 = (const float*)d_in[9];
    const float* gow1 = (const float*)d_in[10];
    const float* gob1 = (const float*)d_in[11];
    const float* gow2 = (const float*)d_in[12];
    const float* gob2 = (const float*)d_in[13];
    const float* dw1  = (const float*)d_in[14];
    const float* db1  = (const float*)d_in[15];
    const float* dw2  = (const float*)d_in[16];
    const float* db2  = (const float*)d_in[17];
    float* out = (float*)d_out;

    dim3 grid(B_ROWS / (TILE * WAVES));   // 4096 blocks, 8 waves * 16 rows each
    dim3 block(256);
    neuralmjp_fused<<<grid, block, SMEM_TOTAL, stream>>>(
        h, u, qw1, qb1, qw2, qb2, giw1, gib1, giw2, gib2,
        gow1, gob1, gow2, gob2, dw1, db1, dw2, db2, out);
}
